// MambaBlock_86912958202093
// MI455X (gfx1250) — compile-verified, matrix operand order checked by reading, not run on hardware
//
#include <hip/hip_runtime.h>
#include <hip/hip_bf16.h>

typedef __attribute__((ext_vector_type(16))) _Float16 v16h;
typedef __attribute__((ext_vector_type(8)))  _Float16 v8h;
typedef __attribute__((ext_vector_type(8)))  float    v8f;

#define B_DIM 2
#define L_DIM 2048
#define DM    1024
#define DI    2048
#define DST   16
#define DTR   64
#define NTOK  (B_DIM * L_DIM)   /* 4096 tokens */
#define NXZ   (2 * DI)          /* 4096 */
#define XD    (DTR + 2 * DST)   /* 96 */
#define XDPAD 128               /* x_proj N padded to 128 so GEMM is branch-free */
#define NCH   32                /* chunks per batch for the parallel scan */
#define LCH   (L_DIM / NCH)     /* 64 steps per chunk */

// ---------------------------------------------------------------------------
// Weight transpose + cast: src[R][C] f32 (row-major) -> dst[Cpad][R] f16,
// rows C..Cpad-1 zero-filled (lets the GEMM run without column guards).
// ---------------------------------------------------------------------------
__global__ void transpose_cast_f16(const float* __restrict__ src,
                                   _Float16* __restrict__ dst,
                                   int R, int C, int Cpad) {
  __shared__ _Float16 tile[32][33];
  int c0 = blockIdx.x * 32, r0 = blockIdx.y * 32;
  int tx = threadIdx.x, ty = threadIdx.y;   // block (32, 8)
  #pragma unroll
  for (int j = 0; j < 32; j += 8) {
    int rr = r0 + ty + j, cc = c0 + tx;
    tile[ty + j][tx] = (rr < R && cc < C) ? (_Float16)src[(size_t)rr * C + cc]
                                          : (_Float16)0.f;
  }
  __syncthreads();
  #pragma unroll
  for (int j = 0; j < 32; j += 8) {
    int rr = r0 + tx, cc = c0 + ty + j;
    if (cc < Cpad && rr < R) dst[(size_t)cc * R + rr] = tile[tx][ty + j];
  }
}

// ---------------------------------------------------------------------------
// LayerNorm over D, output f16 (feeds WMMA GEMM). One 256-thread block / row.
// ---------------------------------------------------------------------------
__global__ void layernorm_f16(const float* __restrict__ x,
                              const float* __restrict__ w,
                              const float* __restrict__ b,
                              _Float16* __restrict__ out, int D) {
  __shared__ float red0[8], red1[8];
  size_t row = blockIdx.x;
  const float* xr = x + row * D;
  float s = 0.f, ss = 0.f;
  for (int i = threadIdx.x; i < D; i += 256) { float v = xr[i]; s += v; ss += v * v; }
  #pragma unroll
  for (int off = 16; off > 0; off >>= 1) {
    s  += __shfl_down(s,  off, 32);
    ss += __shfl_down(ss, off, 32);
  }
  int wid = threadIdx.x >> 5, lid = threadIdx.x & 31;
  if (lid == 0) { red0[wid] = s; red1[wid] = ss; }
  __syncthreads();
  if (threadIdx.x == 0) {
    float ts = 0.f, tss = 0.f;
    for (int i = 0; i < 8; ++i) { ts += red0[i]; tss += red1[i]; }
    red0[0] = ts; red1[0] = tss;
  }
  __syncthreads();
  float mu   = red0[0] / (float)D;
  float var  = red1[0] / (float)D - mu * mu;
  float rsig = rsqrtf(var + 1e-5f);
  for (int i = threadIdx.x; i < D; i += 256)
    out[row * D + i] = (_Float16)((xr[i] - mu) * rsig * w[i] + b[i]);
}

// ---------------------------------------------------------------------------
// WMMA GEMM: out[M,N] = act(A[M,K] @ W[K,N] + bias) (+ residual)
//   A  : f16 row-major, leading dim lda
//   WT : f16, W transposed -> [Npad][K] row-major, Npad = 128*ceil -> the
//        K-loop is completely branch-free: every wave does 4 subtiles.
// Block = 8 waves arranged 4(M) x 2(N); wave computes 16x64 (one A fragment
// reused across 4 B fragments / 4 accumulators) -> 2.5 b128 loads per WMMA.
// act: 0 = none, 1 = hardtanh, 2 = softplus
// ---------------------------------------------------------------------------
__global__ void gemm_f16_wmma(const _Float16* __restrict__ A, int lda,
                              const _Float16* __restrict__ WT,
                              const float* __restrict__ bias,
                              const float* __restrict__ res,
                              float* __restrict__ outF,
                              _Float16* __restrict__ outH,
                              int M, int N, int K, int act) {
  int wave = threadIdx.x >> 5;
  int lane = threadIdx.x & 31;
  int wm = wave >> 1, wn = wave & 1;
  int row0 = blockIdx.y * 64 + wm * 16;
  int col0 = blockIdx.x * 128 + wn * 64;

  int r = lane & 15;            // A: row in tile / B: col in subtile
  int g = lane >> 4;            // K half-group select
  const _Float16* aRow  = A  + (size_t)(row0 + r) * lda + g * 8;
  const _Float16* bRow0 = WT + (size_t)(col0 + r) * K   + g * 8;

  v8f acc[4] = {};
  for (int k0 = 0; k0 < K; k0 += 32) {
    __builtin_prefetch(aRow + k0 + 128, 0, 1);
    __builtin_prefetch(bRow0 + k0 + 128, 0, 1);
    v8h a0 = *(const v8h*)(aRow + k0);
    v8h a1 = *(const v8h*)(aRow + k0 + 16);
    v16h a;
    #pragma unroll
    for (int i = 0; i < 8; ++i) { a[i] = a0[i]; a[i + 8] = a1[i]; }
    #pragma unroll
    for (int j = 0; j < 4; ++j) {
      const _Float16* bR = bRow0 + (size_t)j * 16 * K;
      v8h b0 = *(const v8h*)(bR + k0);
      v8h b1 = *(const v8h*)(bR + k0 + 16);
      v16h b;
      #pragma unroll
      for (int i = 0; i < 8; ++i) { b[i] = b0[i]; b[i + 8] = b1[i]; }
      acc[j] = __builtin_amdgcn_wmma_f32_16x16x32_f16(
          false, a, false, b, (short)0, acc[j], false, false);
    }
  }

  int cbase = lane & 15;
  #pragma unroll
  for (int j = 0; j < 4; ++j) {
    int cg = col0 + j * 16 + cbase;
    if (cg < N) {               // effectively wave-uniform (N % 16 == 0)
      float bv = bias ? bias[cg] : 0.f;
      #pragma unroll
      for (int i = 0; i < 8; ++i) {
        int rr = row0 + i + g * 8;
        float v = acc[j][i] + bv;
        if (act == 1)      v = fminf(1.f, fmaxf(-1.f, v));
        else if (act == 2) v = (v > 20.f) ? v : log1pf(__expf(v));
        if (res)  v += res[(size_t)rr * N + cg];
        if (outF) outF[(size_t)rr * N + cg] = v;
        if (outH) outH[(size_t)rr * N + cg] = (_Float16)v;
      }
    }
  }
}

// ---------------------------------------------------------------------------
// Depthwise causal conv1d (k=4) + bias + SiLU. x_in = xz[:, 0:DI].
// ---------------------------------------------------------------------------
__global__ void conv_silu(const float* __restrict__ xz,
                          const float* __restrict__ cw,
                          const float* __restrict__ cb,
                          float* __restrict__ u, _Float16* __restrict__ uh) {
  size_t idx = (size_t)blockIdx.x * 256 + threadIdx.x;   // over NTOK*DI
  if (idx >= (size_t)NTOK * DI) return;
  int d = (int)(idx % DI);
  size_t tok = idx / DI;
  int l = (int)(tok % L_DIM);
  long long rowbase = (long long)tok * NXZ + d;
  float acc = cb[d];
  #pragma unroll
  for (int j = 0; j < 4; ++j) {
    int lj = l - 3 + j;
    if (lj >= 0) acc += cw[d * 4 + j] * xz[rowbase + (long long)(j - 3) * NXZ];
  }
  float s = acc / (1.f + __expf(-acc));
  u[idx] = s;
  uh[idx] = (_Float16)s;
}

// ---------------------------------------------------------------------------
// Chunked parallel scan, pass 1: per-chunk local scan from zero state.
// Thread per (b, chunk, d): 64 sequential steps. Writes local y, chunk-end
// state hend[b][c][d][16], and sum-of-delta dsum[b][c][d] (cumulative chunk
// decay = exp(A * sum(delta)) since decays multiply).
// xdbl row layout: [dt(64) | B(16) | C(16)], ld = 96.
// ---------------------------------------------------------------------------
__global__ void scan_local(const float* __restrict__ delta,
                           const float* __restrict__ u,
                           const float* __restrict__ xdbl,
                           const float* __restrict__ A_log,
                           float* __restrict__ y,
                           float* __restrict__ hend,
                           float* __restrict__ dsum) {
  int tid = blockIdx.x * 128 + threadIdx.x;   // B*NCH*DI threads
  int d = tid % DI;
  int c = (tid / DI) % NCH;
  int b = tid / (DI * NCH);
  float Ad[DST];
  #pragma unroll
  for (int s = 0; s < DST; ++s) Ad[s] = -__expf(A_log[d * DST + s]);
  float h[DST];
  #pragma unroll
  for (int s = 0; s < DST; ++s) h[s] = 0.f;

  size_t base = ((size_t)b * L_DIM + (size_t)c * LCH) * DI + d;
  size_t xb   = ((size_t)b * L_DIM + (size_t)c * LCH) * XD;
  float dsu = 0.f;
  for (int t = 0; t < LCH; ++t) {
    float dt = delta[base + (size_t)t * DI];
    float ut = u[base + (size_t)t * DI];
    float du = dt * ut;
    dsu += dt;
    const float* xr = xdbl + xb + (size_t)t * XD;
    float acc = 0.f;
    #pragma unroll
    for (int s = 0; s < DST; ++s) {
      float dA = __expf(dt * Ad[s]);
      h[s] = h[s] * dA + du * xr[DTR + s];
      acc += h[s] * xr[DTR + DST + s];
    }
    y[base + (size_t)t * DI] = acc;
  }
  size_t hoff = (((size_t)b * NCH + c) * DI + d) * DST;
  #pragma unroll
  for (int s = 0; s < DST; ++s) hend[hoff + s] = h[s];
  dsum[((size_t)b * NCH + c) * DI + d] = dsu;
}

// ---------------------------------------------------------------------------
// Pass 2: carry recurrence across the 32 chunks (tiny, 65K-way parallel).
// H_{c+1} = hend_c + exp(A * dsum_c) * H_c ; stores carry-IN per chunk.
// Thread per (b, d, s).
// ---------------------------------------------------------------------------
__global__ void chunk_carry(const float* __restrict__ hend,
                            const float* __restrict__ dsum,
                            const float* __restrict__ A_log,
                            float* __restrict__ Hcar) {
  int tid = blockIdx.x * 256 + threadIdx.x;   // B*DI*DST
  int s = tid % DST;
  int d = (tid / DST) % DI;
  int b = tid / (DST * DI);
  float Ad = -__expf(A_log[d * DST + s]);
  float H = 0.f;
  for (int c = 0; c < NCH; ++c) {
    size_t idx = (((size_t)b * NCH + c) * DI + d) * DST + s;
    Hcar[idx] = H;
    float P = __expf(dsum[((size_t)b * NCH + c) * DI + d] * Ad);
    H = hend[idx] + P * H;
  }
}

// ---------------------------------------------------------------------------
// Pass 3: fixup. state_t = local_h_t + cumdecay_t * H_carry, so
// y_t += C_t . (cumdecay_t * H_carry) with cumdecay recomputed from delta.
// Thread per (b, chunk, d); chunk 0 has zero carry.
// ---------------------------------------------------------------------------
__global__ void scan_fixup(const float* __restrict__ delta,
                           const float* __restrict__ xdbl,
                           const float* __restrict__ A_log,
                           const float* __restrict__ Hcar,
                           float* __restrict__ y) {
  int tid = blockIdx.x * 128 + threadIdx.x;   // B*NCH*DI threads
  int d = tid % DI;
  int c = (tid / DI) % NCH;
  int b = tid / (DI * NCH);
  if (c == 0) return;                         // block-uniform: c fixed per block
  float Ad[DST], Hs[DST], cum[DST];
  size_t hoff = (((size_t)b * NCH + c) * DI + d) * DST;
  #pragma unroll
  for (int s = 0; s < DST; ++s) {
    Ad[s] = -__expf(A_log[d * DST + s]);
    Hs[s] = Hcar[hoff + s];
    cum[s] = 1.f;
  }
  size_t base = ((size_t)b * L_DIM + (size_t)c * LCH) * DI + d;
  size_t xb   = ((size_t)b * L_DIM + (size_t)c * LCH) * XD;
  for (int t = 0; t < LCH; ++t) {
    float dt = delta[base + (size_t)t * DI];
    const float* xr = xdbl + xb + (size_t)t * XD;
    float fy = 0.f;
    #pragma unroll
    for (int s = 0; s < DST; ++s) {
      cum[s] *= __expf(dt * Ad[s]);
      fy += xr[DTR + DST + s] * cum[s] * Hs[s];
    }
    y[base + (size_t)t * DI] += fy;
  }
}

// ---------------------------------------------------------------------------
// y2 = (y + u*D) * silu(z), written f16 for the out-proj GEMM.
// ---------------------------------------------------------------------------
__global__ void fuse_gate(const float* __restrict__ y,
                          const float* __restrict__ u,
                          const float* __restrict__ Dp,
                          const float* __restrict__ xz,
                          _Float16* __restrict__ yh) {
  size_t idx = (size_t)blockIdx.x * 256 + threadIdx.x;   // over NTOK*DI
  if (idx >= (size_t)NTOK * DI) return;
  int d = (int)(idx % DI);
  size_t tok = idx / DI;
  float z = xz[tok * NXZ + DI + d];
  float val = (y[idx] + u[idx] * Dp[d]) * (z / (1.f + __expf(-z)));
  yh[idx] = (_Float16)val;
}

// ---------------------------------------------------------------------------
// Host-side orchestration
// ---------------------------------------------------------------------------
static inline void launch_gemm(const _Float16* A, int lda, const _Float16* WT,
                               const float* bias, const float* res,
                               float* outF, _Float16* outH,
                               int M, int N, int K, int act, hipStream_t s) {
  dim3 grid((N + 127) / 128, (M + 63) / 64);
  gemm_f16_wmma<<<grid, 256, 0, s>>>(A, lda, WT, bias, res, outF, outH, M, N, K, act);
}

static inline void launch_transpose(const float* src, _Float16* dst, int R, int C,
                                    int Cpad, hipStream_t s) {
  dim3 grid((Cpad + 31) / 32, (R + 31) / 32);
  transpose_cast_f16<<<grid, dim3(32, 8), 0, s>>>(src, dst, R, C, Cpad);
}

extern "C" void kernel_launch(void* const* d_in, const int* in_sizes, int n_in,
                              void* d_out, int out_size, void* d_ws, size_t ws_size,
                              hipStream_t stream) {
  const float* x_in    = (const float*)d_in[0];
  const float* ln1_w   = (const float*)d_in[1];
  const float* ln1_b   = (const float*)d_in[2];
  const float* W1      = (const float*)d_in[3];
  const float* b1      = (const float*)d_in[4];
  const float* W2      = (const float*)d_in[5];
  const float* b2      = (const float*)d_in[6];
  const float* ln2_w   = (const float*)d_in[7];
  const float* ln2_b   = (const float*)d_in[8];
  const float* in_proj = (const float*)d_in[9];
  const float* conv_w  = (const float*)d_in[10];
  const float* conv_b  = (const float*)d_in[11];
  const float* x_proj  = (const float*)d_in[12];
  const float* dt_proj = (const float*)d_in[13];
  const float* dt_b    = (const float*)d_in[14];
  const float* A_log   = (const float*)d_in[15];
  const float* D_param = (const float*)d_in[16];
  const float* out_w   = (const float*)d_in[17];
  float* out = (float*)d_out;

  // Bump allocator over d_ws (256B aligned slices).
  char* ws = (char*)d_ws;
  size_t off = 0;
  auto alloc = [&](size_t bytes) -> char* {
    char* p = ws + off;
    off += (bytes + 255) & ~(size_t)255;
    return p;
  };
  // f16 transposed weights [Npad][K]
  _Float16* W1T  = (_Float16*)alloc((size_t)DI * DM * 2);        // [2048][1024]
  _Float16* W2T  = (_Float16*)alloc((size_t)DM * DI * 2);        // [1024][2048]
  _Float16* INPT = (_Float16*)alloc((size_t)NXZ * DM * 2);       // [4096][1024]
  _Float16* XPT  = (_Float16*)alloc((size_t)XDPAD * DI * 2);     // [128][2048], rows 96..127 = 0
  _Float16* DTT  = (_Float16*)alloc((size_t)DI * DTR * 2);       // [2048][64]
  _Float16* OUTT = (_Float16*)alloc((size_t)DM * DI * 2);        // [1024][2048]
  // activations
  _Float16* a_h    = (_Float16*)alloc((size_t)NTOK * DM * 2);    // ln1 out, reused for ln2 out
  _Float16* h2_h   = (_Float16*)alloc((size_t)NTOK * DI * 2);    // gated -> u_h -> y_h (disjoint lifetimes)
  float*    x1     = (float*)alloc((size_t)NTOK * DM * 4);
  float*    xz     = (float*)alloc((size_t)NTOK * NXZ * 4);
  float*    u      = (float*)alloc((size_t)NTOK * DI * 4);
  float*    xdbl   = (float*)alloc((size_t)NTOK * XD * 4);
  _Float16* xdbl_h = (_Float16*)alloc((size_t)NTOK * XD * 2);
  float*    delta  = (float*)alloc((size_t)NTOK * DI * 4);
  float*    y      = (float*)alloc((size_t)NTOK * DI * 4);
  // chunked-scan scratch
  float*    hend   = (float*)alloc((size_t)B_DIM * NCH * DI * DST * 4);
  float*    dsum   = (float*)alloc((size_t)B_DIM * NCH * DI * 4);
  float*    Hcar   = (float*)alloc((size_t)B_DIM * NCH * DI * DST * 4);
  (void)ws_size; (void)in_sizes; (void)n_in; (void)out_size;

  // Weight prep (cast + transpose to [Npad][K] f16; XPT zero-padded to 128)
  launch_transpose(W1,      W1T,  DM,  DI,  DI,    stream);
  launch_transpose(W2,      W2T,  DI,  DM,  DM,    stream);
  launch_transpose(in_proj, INPT, DM,  NXZ, NXZ,   stream);
  launch_transpose(x_proj,  XPT,  DI,  XD,  XDPAD, stream);
  launch_transpose(dt_proj, DTT,  DTR, DI,  DI,    stream);
  launch_transpose(out_w,   OUTT, DI,  DM,  DM,    stream);

  // 1) ln1(x) -> a_h
  layernorm_f16<<<NTOK, 256, 0, stream>>>(x_in, ln1_w, ln1_b, a_h, DM);
  // 2) gated = hardtanh(a_h @ W1 + b1) -> h2_h (f16)
  launch_gemm(a_h, DM, W1T, b1, nullptr, nullptr, h2_h, NTOK, DI, DM, 1, stream);
  // 3) x1 = x + gated @ W2 + b2
  launch_gemm(h2_h, DI, W2T, b2, x_in, x1, nullptr, NTOK, DM, DI, 0, stream);
  // 4) ln2(x1) -> a_h
  layernorm_f16<<<NTOK, 256, 0, stream>>>(x1, ln2_w, ln2_b, a_h, DM);
  // 5) xz = ln2 @ in_proj
  launch_gemm(a_h, DM, INPT, nullptr, nullptr, xz, nullptr, NTOK, NXZ, DM, 0, stream);
  // 6) u = silu(causal_conv(x_in part of xz) + conv_b); also f16 copy in h2_h
  {
    size_t total = (size_t)NTOK * DI;
    conv_silu<<<(unsigned)((total + 255) / 256), 256, 0, stream>>>(xz, conv_w, conv_b, u, h2_h);
  }
  // 7) x_dbl = u @ x_proj  (f32 for B/C, f16 for dt slice); WT padded to 128 cols
  launch_gemm(h2_h, DI, XPT, nullptr, nullptr, xdbl, xdbl_h, NTOK, XD, DI, 0, stream);
  // 8) delta = softplus(dt @ dt_proj + dt_b); dt = xdbl[:, :64] (lda = 96)
  launch_gemm(xdbl_h, XD, DTT, dt_b, nullptr, delta, nullptr, NTOK, DI, DTR, 2, stream);
  // 9) chunked selective scan -> y  (local scans || carry recurrence || fixup)
  scan_local<<<(B_DIM * NCH * DI) / 128, 128, 0, stream>>>(delta, u, xdbl, A_log,
                                                           y, hend, dsum);
  chunk_carry<<<(B_DIM * DI * DST) / 256, 256, 0, stream>>>(hend, dsum, A_log, Hcar);
  scan_fixup<<<(B_DIM * NCH * DI) / 128, 128, 0, stream>>>(delta, xdbl, A_log, Hcar, y);
  // 10) y2 = (y + u*D) * silu(z) -> h2_h (f16)
  {
    size_t total = (size_t)NTOK * DI;
    fuse_gate<<<(unsigned)((total + 255) / 256), 256, 0, stream>>>(y, u, D_param, xz, h2_h);
  }
  // 11) out = x1 + y2 @ out_proj
  launch_gemm(h2_h, DI, OUTT, nullptr, x1, out, nullptr, NTOK, DM, DI, 0, stream);
}
